// MeshGraphEncoder_75359496175668
// MI455X (gfx1250) — matrix operand order repaired and checked by reading, not compiled, not run on hardware
//
#include <hip/hip_runtime.h>

typedef _Float16 half_t;
typedef __attribute__((ext_vector_type(16))) _Float16 v16h;
typedef __attribute__((ext_vector_type(8)))  _Float16 v8h;
typedef __attribute__((ext_vector_type(4)))  _Float16 v4h;
typedef __attribute__((ext_vector_type(8)))  float    v8f;

#define DDIM 256
#define NGRID 80000
#define NMESH 20000
#define NEDGE 160000

__device__ __forceinline__ v8f wmma16(v16h a, v16h b, v8f c) {
  // D = A(16x32 f16) x B(32x16 f16) + C(16x16 f32)
  return __builtin_amdgcn_wmma_f32_16x16x32_f16(
      /*neg_a=*/false, a, /*neg_b=*/false, b,
      /*c_mod=*/(short)0, c, /*reuse_a=*/false, /*reuse_b=*/false);
}

// A fragment (ISA 16-bit A 16x32 layout): elems 0-7 -> K = hw*8 + 0..7,
// elems 8-15 -> K = 16 + hw*8 + 0..7.  p = &X[m*ldx + hw*8 + k0].
__device__ __forceinline__ v16h load_a_frag(const half_t* __restrict__ p) {
  v8h lo = *(const v8h*)(p);
  v8h hi = *(const v8h*)(p + 16);
  v16h a;
#pragma unroll
  for (int j = 0; j < 8; ++j) { a[j] = lo[j]; a[8 + j] = hi[j]; }
  return a;
}

// B fragment (same K order as A): elems 0-7 -> K = hw*8 + 0..7,
// elems 8-15 -> K = 16 + hw*8 + 0..7.  p = &Wt[(n)*K + hw*8 + k0].
__device__ __forceinline__ v16h load_b_frag(const half_t* __restrict__ p) {
  v8h lo = *(const v8h*)(p);
  v8h hi = *(const v8h*)(p + 16);
  v16h b;
#pragma unroll
  for (int j = 0; j < 8; ++j) { b[j] = lo[j]; b[8 + j] = hi[j]; }
  return b;
}

// SiLU with hardware v_rcp_f32 (~1 ulp) instead of IEEE divide expansion;
// operands are f16-quantized so this is precision-neutral.
__device__ __forceinline__ float fast_silu(float v) {
  return v * __builtin_amdgcn_rcpf(1.0f + __expf(-v));
}

// X: LDS f16 [16][ldx] tile (rows = M).  Wt: global f16 [256][K] (transposed:
// row n holds K contiguous values).  bias: fp32 [256].  K % 32 == 0 (runtime
// arg => k-loop is not fully unrollable).
//
// Register blocking sized to the compiler's ~128-VGPR budget: N is processed
// in two blocks of 128 columns, keeping 8 v8f accumulators (64 VGPRs) live.
// Each k-step loads ONE A fragment from LDS and streams 8 B fragments from
// the L2-resident transposed weights at constant immediate offsets
// (t*32*K bytes, within the 24-bit IOFFSET range).  The 8 WMMAs per k-step
// are independent chains (no D->A/B RAW hazards, no spills).
template <bool SILU_TO_H>
__device__ __forceinline__ void gemm16_f16(
    const half_t* __restrict__ X, int ldx,
    const half_t* __restrict__ Wt, const float* __restrict__ bias, int K,
    half_t* __restrict__ OutH, float* __restrict__ OutF, int ldo)
{
  const int lane = threadIdx.x & 31;
  const int m    = lane & 15;   // M for A / N for B / N for C-D
  const int hw   = lane >> 4;   // half-wave
  const half_t* xrow = X + m * ldx + hw * 8;

#pragma unroll 1
  for (int nb = 0; nb < 2; ++nb) {            // N-block of 128 columns
    const int nbase = nb * 128;
    const half_t* wbase = Wt + (size_t)(nbase + m) * K + hw * 8;

    v8f acc[8];
#pragma unroll
    for (int t = 0; t < 8; ++t) {
      const float bv = bias[nbase + t * 16 + m];
      acc[t] = (v8f){bv, bv, bv, bv, bv, bv, bv, bv};
    }

#pragma unroll 1
    for (int k0 = 0; k0 < K; k0 += 32) {
      const v16h a = load_a_frag(xrow + k0);
      const half_t* wk = wbase + k0;
#pragma unroll
      for (int t = 0; t < 8; ++t) {
        const v16h b = load_b_frag(wk + (size_t)t * 16 * K);
        acc[t] = wmma16(a, b, acc[t]);
      }
    }

    // C/D layout: vgpr r, lanes 0-15 -> M=r, lanes 16-31 -> M=8+r; N=lane&15
#pragma unroll
    for (int t = 0; t < 8; ++t) {
#pragma unroll
      for (int r = 0; r < 8; ++r) {
        const int row = r + hw * 8;
        const int col = nbase + t * 16 + m;
        if constexpr (SILU_TO_H) {
          OutH[row * ldo + col] = (half_t)fast_silu(acc[t][r]);
        } else {
          OutF[row * ldo + col] = acc[t][r];
        }
      }
    }
  }
}

// Per-lane LayerNorm stats over a 16x256 fp32 LDS tile: lane handles half a
// row (128 cols, float4 LDS reads), halves combined with a wave32 xor-16
// shuffle.
__device__ __forceinline__ void ln_stats(const float* __restrict__ yr,
                                         float& mu, float& rstd) {
  float s = 0.f, ss = 0.f;
#pragma unroll
  for (int c = 0; c < 128; c += 4) {
    const float4 v = *(const float4*)(yr + c);
    s  += v.x + v.y + v.z + v.w;
    ss += v.x * v.x + v.y * v.y + v.z * v.z + v.w * v.w;
  }
  s  += __shfl_xor(s, 16, 32);
  ss += __shfl_xor(ss, 16, 32);
  mu = s * (1.0f / 256.0f);
  const float var = ss * (1.0f / 256.0f) - mu * mu;
  rstd = rsqrtf(var + 1e-5f);
}

// ---------------- utility kernels ----------------

__global__ void zero_f32_kernel(float* __restrict__ p, int n4) {
  int i = blockIdx.x * blockDim.x + threadIdx.x;
  if (i < n4) ((float4*)p)[i] = make_float4(0.f, 0.f, 0.f, 0.f);
}

// W [K][256] fp32 -> Wt [256][K] f16
__global__ void prep_weight_kernel(const float* __restrict__ W,
                                   half_t* __restrict__ Wt, int K) {
  int idx = blockIdx.x * blockDim.x + threadIdx.x;
  if (idx < K * 256) {
    int n = idx / K, k = idx - n * K;
    Wt[idx] = (half_t)W[(size_t)k * 256 + n];
  }
}

// ---------------- edge MLP + scatter ----------------

__global__ void __attribute__((amdgpu_flat_work_group_size(32, 32)))
edge_mlp_scatter_kernel(
    const float* __restrict__ efeat, const float* __restrict__ gridf,
    const float* __restrict__ meshf, const int* __restrict__ src_idx,
    const int* __restrict__ dst_idx,
    const half_t* __restrict__ W1t, const float* __restrict__ b1,
    const half_t* __restrict__ W2t, const float* __restrict__ b2,
    const float* __restrict__ g, const float* __restrict__ beta,
    float* __restrict__ agg)
{
  constexpr int LDX = 776;   // 768 + 8 pad (halves)
  constexpr int LDH = 264;   // 256 + 8 pad
  extern __shared__ char smem[];
  half_t* x = (half_t*)smem;                                   // 16*776*2
  half_t* h = (half_t*)(smem + 16 * LDX * sizeof(half_t));     // 16*264*2
  float*  y = (float*)(smem + 16 * LDX * sizeof(half_t)
                            + 16 * LDH * sizeof(half_t));      // 16*264*4
  __shared__ int sidx[16], didx[16];

  const int lane = threadIdx.x;
  const int e0 = blockIdx.x * 16;
  if (lane < 16) {
    sidx[lane] = src_idx[e0 + lane];
    didx[lane] = dst_idx[e0 + lane];
  }
  __syncthreads();

  // gather [efeat | grid[src] | mesh[dst]] -> f16 LDS tile, float4 granules
  for (int i = lane; i < 16 * 192; i += 32) {
    const int r  = i / 192;
    const int c4 = (i - r * 192) * 4;
    const float* sp;
    if (c4 < 256)      sp = efeat + (size_t)(e0 + r) * DDIM + c4;
    else if (c4 < 512) sp = gridf + (size_t)sidx[r] * DDIM + (c4 - 256);
    else               sp = meshf + (size_t)didx[r] * DDIM + (c4 - 512);
    const float4 v = *(const float4*)sp;
    v4h hv = {(half_t)v.x, (half_t)v.y, (half_t)v.z, (half_t)v.w};
    *(v4h*)(x + r * LDX + c4) = hv;
  }
  __syncthreads();

  gemm16_f16<true >(x, LDX, W1t, b1, 768, h, nullptr, LDH);
  __syncthreads();
  gemm16_f16<false>(h, LDH, W2t, b2, 256, nullptr, y, LDH);
  __syncthreads();

  const int m = lane & 15, hw = lane >> 4;
  const float* yr = y + m * LDH + hw * 128;
  float mu, rstd;
  ln_stats(yr, mu, rstd);

  float* ap       = agg  + (size_t)didx[m] * DDIM + hw * 128;
  const float* gp = g    + hw * 128;
  const float* bp = beta + hw * 128;
  for (int c = 0; c < 128; c += 4) {
    const float4 v  = *(const float4*)(yr + c);
    const float4 gv = *(const float4*)(gp + c);
    const float4 bv = *(const float4*)(bp + c);
    atomicAdd(ap + c + 0, (v.x - mu) * rstd * gv.x + bv.x);
    atomicAdd(ap + c + 1, (v.y - mu) * rstd * gv.y + bv.y);
    atomicAdd(ap + c + 2, (v.z - mu) * rstd * gv.z + bv.z);
    atomicAdd(ap + c + 3, (v.w - mu) * rstd * gv.w + bv.w);
  }
}

// ---------------- grid (src) node MLP: D -> H -> D, residual ----------------

__global__ void __attribute__((amdgpu_flat_work_group_size(32, 32)))
grid_node_mlp_kernel(
    const float* __restrict__ xin,
    const half_t* __restrict__ W1t, const float* __restrict__ b1,
    const half_t* __restrict__ W2t, const float* __restrict__ b2,
    const float* __restrict__ g, const float* __restrict__ beta,
    float* __restrict__ out)
{
  constexpr int LDX = 264, LDH = 264;
  extern __shared__ char smem[];
  half_t* x = (half_t*)smem;
  half_t* h = (half_t*)(smem + 16 * LDX * sizeof(half_t));
  float*  y = (float*)(smem + 16 * LDX * sizeof(half_t)
                            + 16 * LDH * sizeof(half_t));
  const int lane = threadIdx.x;
  const size_t r0 = (size_t)blockIdx.x * 16;

  for (int i = lane; i < 16 * 64; i += 32) {
    const int r = i >> 6, c4 = (i & 63) * 4;
    const float4 v = *(const float4*)(xin + (r0 + r) * DDIM + c4);
    v4h hv = {(half_t)v.x, (half_t)v.y, (half_t)v.z, (half_t)v.w};
    *(v4h*)(x + r * LDX + c4) = hv;
  }
  __syncthreads();

  gemm16_f16<true >(x, LDX, W1t, b1, 256, h, nullptr, LDH);
  __syncthreads();
  gemm16_f16<false>(h, LDH, W2t, b2, 256, nullptr, y, LDH);
  __syncthreads();

  const int m = lane & 15, hw = lane >> 4;
  const float* yr = y + m * LDH + hw * 128;
  float mu, rstd;
  ln_stats(yr, mu, rstd);

  const float* xr = xin + (r0 + m) * DDIM + hw * 128;
  float* op       = out + (r0 + m) * DDIM + hw * 128;
  const float* gp = g    + hw * 128;
  const float* bp = beta + hw * 128;
  for (int c = 0; c < 128; c += 4) {
    const float4 v  = *(const float4*)(yr + c);
    const float4 gv = *(const float4*)(gp + c);
    const float4 bv = *(const float4*)(bp + c);
    const float4 xv = *(const float4*)(xr + c);
    float4 o;
    o.x = xv.x + (v.x - mu) * rstd * gv.x + bv.x;
    o.y = xv.y + (v.y - mu) * rstd * gv.y + bv.y;
    o.z = xv.z + (v.z - mu) * rstd * gv.z + bv.z;
    o.w = xv.w + (v.w - mu) * rstd * gv.w + bv.w;
    *(float4*)(op + c) = o;
  }
}

// ---------------- mesh (dst) node MLP: [agg|mesh] 2D -> H -> D, residual ----

__global__ void __attribute__((amdgpu_flat_work_group_size(32, 32)))
mesh_node_mlp_kernel(
    const float* __restrict__ agg, const float* __restrict__ meshf,
    const half_t* __restrict__ W1t, const float* __restrict__ b1,
    const half_t* __restrict__ W2t, const float* __restrict__ b2,
    const float* __restrict__ g, const float* __restrict__ beta,
    float* __restrict__ out)
{
  constexpr int LDX = 520, LDH = 264;   // 512 + 8 pad
  extern __shared__ char smem[];
  half_t* x = (half_t*)smem;
  half_t* h = (half_t*)(smem + 16 * LDX * sizeof(half_t));
  float*  y = (float*)(smem + 16 * LDX * sizeof(half_t)
                            + 16 * LDH * sizeof(half_t));
  const int lane = threadIdx.x;
  const size_t r0 = (size_t)blockIdx.x * 16;

  for (int i = lane; i < 16 * 128; i += 32) {
    const int r = i >> 7, c4 = (i & 127) * 4;
    const float* sp = (c4 < 256)
        ? (agg   + (r0 + r) * DDIM + c4)
        : (meshf + (r0 + r) * DDIM + (c4 - 256));
    const float4 v = *(const float4*)sp;
    v4h hv = {(half_t)v.x, (half_t)v.y, (half_t)v.z, (half_t)v.w};
    *(v4h*)(x + r * LDX + c4) = hv;
  }
  __syncthreads();

  gemm16_f16<true >(x, LDX, W1t, b1, 512, h, nullptr, LDH);
  __syncthreads();
  gemm16_f16<false>(h, LDH, W2t, b2, 256, nullptr, y, LDH);
  __syncthreads();

  const int m = lane & 15, hw = lane >> 4;
  const float* yr = y + m * LDH + hw * 128;
  float mu, rstd;
  ln_stats(yr, mu, rstd);

  const float* xr = meshf + (r0 + m) * DDIM + hw * 128;
  float* op       = out   + (r0 + m) * DDIM + hw * 128;
  const float* gp = g    + hw * 128;
  const float* bp = beta + hw * 128;
  for (int c = 0; c < 128; c += 4) {
    const float4 v  = *(const float4*)(yr + c);
    const float4 gv = *(const float4*)(gp + c);
    const float4 bv = *(const float4*)(bp + c);
    const float4 xv = *(const float4*)(xr + c);
    float4 o;
    o.x = xv.x + (v.x - mu) * rstd * gv.x + bv.x;
    o.y = xv.y + (v.y - mu) * rstd * gv.y + bv.y;
    o.z = xv.z + (v.z - mu) * rstd * gv.z + bv.z;
    o.w = xv.w + (v.w - mu) * rstd * gv.w + bv.w;
    *(float4*)(op + c) = o;
  }
}

// ---------------- host launch ----------------

extern "C" void kernel_launch(void* const* d_in, const int* in_sizes, int n_in,
                              void* d_out, int out_size, void* d_ws, size_t ws_size,
                              hipStream_t stream) {
  (void)in_sizes; (void)n_in; (void)out_size; (void)ws_size;

  const float* efeat = (const float*)d_in[0];
  const float* gridf = (const float*)d_in[1];
  const float* meshf = (const float*)d_in[2];
  const int*   src   = (const int*)d_in[3];
  const int*   dst   = (const int*)d_in[4];
  const float* eW1 = (const float*)d_in[5],  *eb1 = (const float*)d_in[6];
  const float* eW2 = (const float*)d_in[7],  *eb2 = (const float*)d_in[8];
  const float* eg  = (const float*)d_in[9],  *ebe = (const float*)d_in[10];
  const float* sW1 = (const float*)d_in[11], *sb1 = (const float*)d_in[12];
  const float* sW2 = (const float*)d_in[13], *sb2 = (const float*)d_in[14];
  const float* sg  = (const float*)d_in[15], *sbe = (const float*)d_in[16];
  const float* dW1 = (const float*)d_in[17], *db1 = (const float*)d_in[18];
  const float* dW2 = (const float*)d_in[19], *db2 = (const float*)d_in[20];
  const float* dg  = (const float*)d_in[21], *dbe = (const float*)d_in[22];

  float* out      = (float*)d_out;
  float* grid_out = out;
  float* mesh_out = out + (size_t)NGRID * DDIM;

  // workspace layout: agg [NM*D f32] then six transposed f16 weights
  char* ws = (char*)d_ws;
  float*  agg  = (float*)ws;                 size_t off = (size_t)NMESH * DDIM * 4;
  half_t* eW1t = (half_t*)(ws + off);        off += (size_t)768 * 256 * 2;
  half_t* eW2t = (half_t*)(ws + off);        off += (size_t)256 * 256 * 2;
  half_t* sW1t = (half_t*)(ws + off);        off += (size_t)256 * 256 * 2;
  half_t* sW2t = (half_t*)(ws + off);        off += (size_t)256 * 256 * 2;
  half_t* dW1t = (half_t*)(ws + off);        off += (size_t)512 * 256 * 2;
  half_t* dW2t = (half_t*)(ws + off);        off += (size_t)256 * 256 * 2;

  // zero the scatter accumulator (deterministic per call)
  {
    int n4 = NMESH * DDIM / 4;
    zero_f32_kernel<<<(n4 + 255) / 256, 256, 0, stream>>>(agg, n4);
  }
  // stage transposed f16 weights (1.2 MB, L2-resident for the whole launch)
  prep_weight_kernel<<<(768 * 256 + 255) / 256, 256, 0, stream>>>(eW1, eW1t, 768);
  prep_weight_kernel<<<(256 * 256 + 255) / 256, 256, 0, stream>>>(eW2, eW2t, 256);
  prep_weight_kernel<<<(256 * 256 + 255) / 256, 256, 0, stream>>>(sW1, sW1t, 256);
  prep_weight_kernel<<<(256 * 256 + 255) / 256, 256, 0, stream>>>(sW2, sW2t, 256);
  prep_weight_kernel<<<(512 * 256 + 255) / 256, 256, 0, stream>>>(dW1, dW1t, 512);
  prep_weight_kernel<<<(256 * 256 + 255) / 256, 256, 0, stream>>>(dW2, dW2t, 256);

  // edge MLP + scatter-sum: 10000 tiles of 16 edges, one wave each
  {
    size_t smem = 16 * 776 * sizeof(half_t) + 16 * 264 * sizeof(half_t)
                + 16 * 264 * sizeof(float);
    edge_mlp_scatter_kernel<<<NEDGE / 16, 32, smem, stream>>>(
        efeat, gridf, meshf, src, dst, eW1t, eb1, eW2t, eb2, eg, ebe, agg);
  }
  // grid node MLP (independent of scatter)
  {
    size_t smem = 16 * 264 * sizeof(half_t) + 16 * 264 * sizeof(half_t)
                + 16 * 264 * sizeof(float);
    grid_node_mlp_kernel<<<NGRID / 16, 32, smem, stream>>>(
        gridf, sW1t, sb1, sW2t, sb2, sg, sbe, grid_out);
  }
  // mesh node MLP (consumes agg; same stream => ordered after edge kernel)
  {
    size_t smem = 16 * 520 * sizeof(half_t) + 16 * 264 * sizeof(half_t)
                + 16 * 264 * sizeof(float);
    mesh_node_mlp_kernel<<<NMESH / 16, 32, smem, stream>>>(
        agg, meshf, dW1t, db1, dW2t, db2, dg, dbe, mesh_out);
  }
}
